// AutoEncoderTopK_12249246728713
// MI455X (gfx1250) — matrix operand order repaired and checked by reading, not compiled, not run on hardware
//
#include <hip/hip_runtime.h>

// ---------------- Problem constants ----------------
#define B_      8192
#define D_      768
#define F_      32768
#define K_      32
#define ROWS    16          // rows (batch) per block in encode kernel
#define FTILE   128         // f-columns per pass (8 waves x 16)
#define FCHUNK  2048        // f-columns per block (16 passes)
#define NPASS   (FCHUNK / FTILE)
#define DCHUNK  32          // d staged per iteration (one WMMA K-step)
#define NITER   (D_ / DCHUNK)            // 24
#define NCHUNKS (F_ / FCHUNK)            // 16 candidate chunks per row
#define CANDS   (NCHUNKS * K_)           // 512 candidates per row

typedef __bf16   bf16_t;
typedef bf16_t   v16bf __attribute__((ext_vector_type(16)));
typedef float    v8f   __attribute__((ext_vector_type(8)));
typedef unsigned int u32x4 __attribute__((ext_vector_type(4)));
typedef int gvec_i32x4 __attribute__((vector_size(16)));   // matches builtin param type

// ---- CDNA5 async global->LDS path (guarded; falls back to sync copy) ----
#if defined(__has_builtin)
#  if __has_builtin(__builtin_amdgcn_global_load_async_to_lds_b128) && \
      __has_builtin(__builtin_amdgcn_s_wait_asynccnt)
#    define USE_ASYNC_LDS 1
#  endif
#endif
#ifndef USE_ASYNC_LDS
#  define USE_ASYNC_LDS 0
#endif

__device__ __forceinline__ void async_cp128(void* lds, const void* g) {
#if USE_ASYNC_LDS
    __builtin_amdgcn_global_load_async_to_lds_b128(
        (gvec_i32x4*)(g), (gvec_i32x4*)(lds), 0, 0);
#else
    *(u32x4*)lds = *(const u32x4*)g;
#endif
}
__device__ __forceinline__ void wait_async() {
#if USE_ASYNC_LDS
    __builtin_amdgcn_s_wait_asynccnt(0);
#endif
}

__device__ __forceinline__ unsigned short f2bf(float f) {
    unsigned u = __float_as_uint(f);
    u += 0x7FFFu + ((u >> 16) & 1u);          // round-to-nearest-even
    return (unsigned short)(u >> 16);
}
__device__ __forceinline__ float bf2f(unsigned short h) {
    return __uint_as_float(((unsigned)h) << 16);
}

// -------- Kernel 1a: W_enc fp32 -> bf16 hi/lo split --------
__global__ void convert_w_kernel(const float* __restrict__ W,
                                 unsigned short* __restrict__ Whi,
                                 unsigned short* __restrict__ Wlo, int n) {
    for (int i = blockIdx.x * blockDim.x + threadIdx.x; i < n;
         i += gridDim.x * blockDim.x) {
        float w = W[i];
        unsigned short hi = f2bf(w);
        float lo = w - bf2f(hi);
        Whi[i] = hi;
        Wlo[i] = f2bf(lo);
    }
}

// -------- Kernel 1b: (x - b_dec) fp32 -> bf16 hi/lo split --------
__global__ void convert_x_kernel(const float* __restrict__ x,
                                 const float* __restrict__ b_dec,
                                 unsigned short* __restrict__ Xhi,
                                 unsigned short* __restrict__ Xlo, int n) {
    for (int i = blockIdx.x * blockDim.x + threadIdx.x; i < n;
         i += gridDim.x * blockDim.x) {
        float v = x[i] - b_dec[i % D_];
        unsigned short hi = f2bf(v);
        float lo = v - bf2f(hi);
        Xhi[i] = hi;
        Xlo[i] = f2bf(lo);
    }
}

// -------- Kernel 2: fused bf16x3 WMMA encoder + ReLU + running top-32 --------
// grid: (B/16, F/2048), block: 256 threads (8 waves of 32)
// Double-buffered async global->LDS staging overlapped with WMMA compute.
__global__ __launch_bounds__(256) void encode_topk_kernel(
    const unsigned short* __restrict__ Xhi, const unsigned short* __restrict__ Xlo,
    const unsigned short* __restrict__ Whi, const unsigned short* __restrict__ Wlo,
    const float* __restrict__ b_enc,
    float* __restrict__ cand_val, int* __restrict__ cand_idx)
{
    __shared__ __attribute__((aligned(16))) unsigned short lxh[2][ROWS * DCHUNK];
    __shared__ __attribute__((aligned(16))) unsigned short lxl[2][ROWS * DCHUNK];
    __shared__ __attribute__((aligned(16))) unsigned short lwh[2][FTILE * DCHUNK];
    __shared__ __attribute__((aligned(16))) unsigned short lwl[2][FTILE * DCHUNK];
    // per-row scratch: [0..31] running top-32, [32..159] new candidates of a pass
    __shared__ float sval[ROWS][160];
    __shared__ int   sidx[ROWS][160];

    const int tid  = threadIdx.x;
    const int wave = tid >> 5;
    const int lane = tid & 31;
    const int m    = lane & 15;       // row-in-tile for A, n-in-tile for B
    const int half = lane >> 4;
    const int rowbase    = blockIdx.x * ROWS;
    const int fchunkbase = blockIdx.y * FCHUNK;

    // staging coordinates: W tile 128x32 -> 16 bf16 per thread (2x b128)
    const int wfr = tid >> 1;            // f-row 0..127
    const int wdd = (tid & 1) * 16;      // 0 or 16
    // X tile 16x32 -> first 64 threads, 8 bf16 (1x b128)
    const int xr  = tid >> 2;
    const int xdd = (tid & 3) * 8;

    // init running top-32 with -1 sentinels
    for (int i = tid; i < ROWS * 32; i += 256) {
        sval[i >> 5][i & 31] = -1.0f;
        sidx[i >> 5][i & 31] = 0;
    }

    for (int pass = 0; pass < NPASS; ++pass) {
        const int ftb = fchunkbase + pass * FTILE;
        v8f acc = {0.f, 0.f, 0.f, 0.f, 0.f, 0.f, 0.f, 0.f};

        // issue one d-chunk of async copies into LDS buffer `buf`
        auto stage = [&](int buf, int d0) {
            const size_t gw = (size_t)(ftb + wfr) * D_ + d0 + wdd;
            async_cp128(&lwh[buf][wfr * DCHUNK + wdd],     &Whi[gw]);
            async_cp128(&lwh[buf][wfr * DCHUNK + wdd + 8], &Whi[gw + 8]);
            async_cp128(&lwl[buf][wfr * DCHUNK + wdd],     &Wlo[gw]);
            async_cp128(&lwl[buf][wfr * DCHUNK + wdd + 8], &Wlo[gw + 8]);
            if (d0 + 2 * DCHUNK < D_)
                __builtin_prefetch(&Whi[gw + 2 * DCHUNK], 0, 3);
            if (tid < 64) {
                const size_t gx = (size_t)(rowbase + xr) * D_ + d0 + xdd;
                async_cp128(&lxh[buf][xr * DCHUNK + xdd], &Xhi[gx]);
                async_cp128(&lxl[buf][xr * DCHUNK + xdd], &Xlo[gx]);
            }
        };

        stage(0, 0);
        wait_async();
        __syncthreads();

        for (int it = 0; it < NITER; ++it) {
            const int buf = it & 1;
            if (it + 1 < NITER) stage(buf ^ 1, (it + 1) * DCHUNK);

            union { v16bf v; u32x4 q[2]; } ah, al, bh, bl;
            // A fragment (16x32 bf16): lane<16 -> K 0..7 & 16..23; lane>=16 -> +8
            const int xoff = m * DCHUNK + half * 8;
            ah.q[0] = *(const u32x4*)&lxh[buf][xoff];
            ah.q[1] = *(const u32x4*)&lxh[buf][xoff + 16];
            al.q[0] = *(const u32x4*)&lxl[buf][xoff];
            al.q[1] = *(const u32x4*)&lxl[buf][xoff + 16];
            // B fragment (32x16 bf16): lane<16 -> K 0..7 & 16..23; lane>=16 -> +8
            const int woff = (wave * 16 + m) * DCHUNK + half * 8;
            bh.q[0] = *(const u32x4*)&lwh[buf][woff];
            bh.q[1] = *(const u32x4*)&lwh[buf][woff + 16];
            bl.q[0] = *(const u32x4*)&lwl[buf][woff];
            bl.q[1] = *(const u32x4*)&lwl[buf][woff + 16];
            // bf16x3 emulated-fp32 product
            acc = __builtin_amdgcn_wmma_f32_16x16x32_bf16(false, ah.v, false, bh.v, (short)0, acc, false, false);
            acc = __builtin_amdgcn_wmma_f32_16x16x32_bf16(false, ah.v, false, bl.v, (short)0, acc, false, false);
            acc = __builtin_amdgcn_wmma_f32_16x16x32_bf16(false, al.v, false, bh.v, (short)0, acc, false, false);

            wait_async();       // next buffer fully resident in LDS
            __syncthreads();    // all waves done reading `buf`, staging visible
        }

        // epilogue: bias + ReLU, write this pass's 128 candidates per row
        const int fg = ftb + wave * 16 + m;
        const float bias = b_enc[fg];
        #pragma unroll
        for (int i = 0; i < 8; ++i) {
            const int r = i + half * 8;            // C layout: M = i + 8*(lane>>4)
            float v = acc[i] + bias;
            v = v > 0.f ? v : 0.f;
            sval[r][32 + wave * 16 + m] = v;
            sidx[r][32 + wave * 16 + m] = fg;
        }
        __syncthreads();

        // running top-32 merge: each wave owns rows {wave, wave+8}
        for (int rr = 0; rr < 2; ++rr) {
            const int r = wave + rr * 8;
            volatile float* vrow = &sval[r][0];
            float keepv = 0.f; int keepi = 0;
            for (int k = 0; k < 32; ++k) {
                float bv = -2.f; int bp = 0;
                for (int e = lane; e < 160; e += 32) {
                    float v = vrow[e];
                    if (v > bv) { bv = v; bp = e; }
                }
                #pragma unroll
                for (int off = 16; off > 0; off >>= 1) {
                    float ov = __shfl_xor(bv, off, 32);
                    int   op = __shfl_xor(bp, off, 32);
                    if (ov > bv || (ov == bv && op < bp)) { bv = ov; bp = op; }
                }
                if (lane == k) { keepv = bv; keepi = sidx[r][bp]; }
                if (lane == 0) vrow[bp] = -1.f;      // destructive mark
            }
            sval[r][lane] = keepv;                   // compact back into [0..31]
            sidx[r][lane] = keepi;
        }
        __syncthreads();
    }

    // write final top-32 per row for this f-chunk
    for (int rr = 0; rr < 2; ++rr) {
        const int r = wave + rr * 8;
        const size_t o = (size_t)(rowbase + r) * CANDS + (size_t)blockIdx.y * 32 + lane;
        cand_val[o] = sval[r][lane];
        cand_idx[o] = sidx[r][lane];
    }
}

// -------- Kernel 3: merge 512 candidates -> top-32, sparse decode --------
// grid: B blocks, 256 threads. Decode uses W_enc rows (== W_dec columns), coalesced.
__global__ __launch_bounds__(256) void merge_decode_kernel(
    const float* __restrict__ cand_val, const int* __restrict__ cand_idx,
    const float* __restrict__ Wenc, const float* __restrict__ b_dec,
    float* __restrict__ out)
{
    __shared__ float cv[CANDS];
    __shared__ int   ci[CANDS];
    __shared__ float selv[K_];
    __shared__ int   seli[K_];
    const int tid = threadIdx.x;
    const int row = blockIdx.x;

    for (int i = tid; i < CANDS; i += 256) {
        cv[i] = cand_val[(size_t)row * CANDS + i];
        ci[i] = cand_idx[(size_t)row * CANDS + i];
    }
    __syncthreads();

    if (tid < 32) {                       // wave 0 does the selection
        volatile float* vcv = &cv[0];
        float keepv = 0.f; int keepi = 0;
        for (int k = 0; k < K_; ++k) {
            float bv = -2.f; int bp = 0;
            for (int e = tid; e < CANDS; e += 32) {
                float v = vcv[e];
                if (v > bv) { bv = v; bp = e; }
            }
            #pragma unroll
            for (int off = 16; off > 0; off >>= 1) {
                float ov = __shfl_xor(bv, off, 32);
                int   op = __shfl_xor(bp, off, 32);
                if (ov > bv || (ov == bv && op < bp)) { bv = ov; bp = op; }
            }
            if (tid == k) { keepv = bv; keepi = ci[bp]; }
            if (tid == 0) vcv[bp] = -1.f;
        }
        selv[tid] = keepv;
        seli[tid] = keepi;
    }
    __syncthreads();

    // decode: 768 = 256 * 3 outputs per row
    float a0 = b_dec[tid], a1 = b_dec[tid + 256], a2 = b_dec[tid + 512];
    for (int k = 0; k < K_; ++k) {
        const float v = selv[k];          // uniform across block
        if (v > 0.f) {
            const float* wr = Wenc + (size_t)seli[k] * D_;
            a0 += v * wr[tid];
            a1 += v * wr[tid + 256];
            a2 += v * wr[tid + 512];
        }
    }
    out[(size_t)row * D_ + tid]       = a0;
    out[(size_t)row * D_ + tid + 256] = a1;
    out[(size_t)row * D_ + tid + 512] = a2;
}

// ---------------- Launch ----------------
extern "C" void kernel_launch(void* const* d_in, const int* in_sizes, int n_in,
                              void* d_out, int out_size, void* d_ws, size_t ws_size,
                              hipStream_t stream) {
    const float* x     = (const float*)d_in[0];
    const float* W_enc = (const float*)d_in[1];
    const float* b_enc = (const float*)d_in[2];
    const float* W_dec = (const float*)d_in[3]; (void)W_dec; // decode via W_enc = W_dec.T
    const float* b_dec = (const float*)d_in[4];
    float* out = (float*)d_out;

    // Workspace layout (~152 MB total)
    char* ws = (char*)d_ws;
    unsigned short* Whi = (unsigned short*)(ws);                 // 50,331,648 B
    unsigned short* Wlo = (unsigned short*)(ws + 50331648);      // 50,331,648 B
    unsigned short* Xhi = (unsigned short*)(ws + 100663296);     // 12,582,912 B
    unsigned short* Xlo = (unsigned short*)(ws + 113246208);     // 12,582,912 B
    float*          cand_val = (float*)(ws + 125829120);         // 16,777,216 B
    int*            cand_idx = (int*)  (ws + 142606336);         // 16,777,216 B

    convert_w_kernel<<<4096, 256, 0, stream>>>(W_enc, Whi, Wlo, F_ * D_);
    convert_x_kernel<<<2048, 256, 0, stream>>>(x, b_dec, Xhi, Xlo, B_ * D_);

    dim3 grid(B_ / ROWS, F_ / FCHUNK);
    encode_topk_kernel<<<grid, 256, 0, stream>>>(Xhi, Xlo, Whi, Wlo, b_enc,
                                                 cand_val, cand_idx);

    merge_decode_kernel<<<B_, 256, 0, stream>>>(cand_val, cand_idx, W_enc, b_dec, out);
}
